// SpatialGatingUnit_multi_9826885173360
// MI455X (gfx1250) — compile-verified, matrix operand order checked by reading, not run on hardware
//
#include <hip/hip_runtime.h>
#include <hip/hip_bf16.h>
#include <math.h>

#define BATCH 32
#define SEQ   512
#define DFF   1536
#define D24   64
#define ROWS  (BATCH * SEQ)   // 16384

typedef __attribute__((ext_vector_type(8)))  __bf16 v8bf;
typedef __attribute__((ext_vector_type(16))) __bf16 v16bf;
typedef __attribute__((ext_vector_type(8)))  float  v8f;

struct __attribute__((aligned(16))) U4 { unsigned int x, y, z, w; };

__device__ __forceinline__ float gelu_f(float v) {
    return 0.5f * v * (1.0f + erff(v * 0.70710678118654752f));
}

__device__ __forceinline__ unsigned short f2bf(float f) {
    unsigned int u = __float_as_uint(f);
    u += 0x7fffu + ((u >> 16) & 1u);   // round-to-nearest-even
    return (unsigned short)(u >> 16);
}

// ---------------- 1) LayerNorm statistics: one block per (b,s) row ----------------
__global__ void k_ln_stats(const float* __restrict__ x,
                           float* __restrict__ mu, float* __restrict__ rs) {
    int row = blockIdx.x;
    const float* xr = x + (size_t)row * DFF;
    float s = 0.f, s2 = 0.f;
    for (int i = threadIdx.x; i < DFF; i += 256) {
        float v = xr[i]; s += v; s2 += v * v;
    }
    __shared__ float sh[256], sh2[256];
    sh[threadIdx.x] = s; sh2[threadIdx.x] = s2;
    __syncthreads();
    for (int off = 128; off > 0; off >>= 1) {
        if (threadIdx.x < off) {
            sh[threadIdx.x]  += sh[threadIdx.x + off];
            sh2[threadIdx.x] += sh2[threadIdx.x + off];
        }
        __syncthreads();
    }
    if (threadIdx.x == 0) {
        float m   = sh[0]  * (1.f / DFF);
        float var = sh2[0] * (1.f / DFF) - m * m;
        mu[row] = m;
        rs[row] = rsqrtf(var + 1e-5f);
    }
}

// ------------- 2) Normalize, bf16-convert, transpose 64x64 tiles -> xnT[b][d][s] -------------
__global__ void k_norm_t(const float* __restrict__ x,
                         const float* __restrict__ mu, const float* __restrict__ rs,
                         const float* __restrict__ lng, const float* __restrict__ lnb,
                         unsigned short* __restrict__ xnT) {
    int b = blockIdx.z, s0 = blockIdx.y * 64, d0 = blockIdx.x * 64;
    __shared__ unsigned short tile[64][66];
    int t = threadIdx.x;

    // load + normalize phase: thread covers 16 consecutive d in one s-row
    int sL = t >> 2, c0 = (t & 3) * 16;
    int row = b * SEQ + s0 + sL;
    const float* xr = x + (size_t)row * DFF + d0 + c0;
    float m = mu[row], r = rs[row];
    #pragma unroll
    for (int k = 0; k < 16; ++k) {
        int d = d0 + c0 + k;
        float v = (xr[k] - m) * r * lng[d] + lnb[d];
        tile[sL][c0 + k] = f2bf(v);
    }
    __syncthreads();

    // store phase: thread covers 16 consecutive s in one d-row (coalesced)
    int dL = t >> 2, sStart = (t & 3) * 16;
    unsigned int w[8];
    #pragma unroll
    for (int i = 0; i < 8; ++i) {
        unsigned int lo = tile[sStart + 2 * i][dL];
        unsigned int hi = tile[sStart + 2 * i + 1][dL];
        w[i] = lo | (hi << 16);
    }
    unsigned short* orow = xnT + ((size_t)b * DFF + d0 + dL) * SEQ + s0 + sStart;
    ((U4*)orow)[0] = U4{w[0], w[1], w[2], w[3]};
    ((U4*)orow)[1] = U4{w[4], w[5], w[6], w[7]};
}

// ---------------- 3) Convert Wi / Wii to bf16 ----------------
__global__ void k_prep(const float* __restrict__ Wi, const float* __restrict__ Wii,
                       unsigned short* __restrict__ WiB, unsigned short* __restrict__ WiiB) {
    int i = blockIdx.x * 256 + threadIdx.x;
    if (i < SEQ * SEQ) {
        WiB[i]  = f2bf(Wi[i]);
        WiiB[i] = f2bf(Wii[i]);
    }
}

// ---------------- 4) Gate stage 1: per-row h[k], scaled by wc[s] ----------------
__global__ void k_gate1(const float* __restrict__ x,
                        const float* __restrict__ mu, const float* __restrict__ rs,
                        const float* __restrict__ lng, const float* __restrict__ lnb,
                        const float* __restrict__ w1, const float* __restrict__ b1,
                        const float* __restrict__ wc, float* __restrict__ hvals) {
    int wave = blockIdx.x * 8 + (threadIdx.x >> 5);   // row id 0..16383
    int lane = threadIdx.x & 31;
    int s = wave & (SEQ - 1);
    const float* xr = x + (size_t)wave * DFF;
    float m = mu[wave], r = rs[wave];
    float c0 = (xr[lane]      - m) * r * lng[lane]      + lnb[lane];
    float c1 = (xr[lane + 32] - m) * r * lng[lane + 32] + lnb[lane + 32];
    float wcs = wc[s];
    #pragma unroll
    for (int k = 0; k < 10; ++k) {
        float p = c0 * w1[lane * 10 + k] + c1 * w1[(lane + 32) * 10 + k];
        for (int off = 16; off > 0; off >>= 1) p += __shfl_xor(p, off, 32);
        if (lane == 0) {
            float hk = gelu_f(p + b1[k]);
            hvals[(size_t)wave * 10 + k] = wcs * hk;
        }
    }
}

// ---------------- 5) Gate stage 2: deterministic reduce over s, sigmoid, round ----------------
__global__ void k_gate2(const float* __restrict__ hvals, const float* __restrict__ bc,
                        const float* __restrict__ w2, const float* __restrict__ b2,
                        int* __restrict__ sel) {
    __shared__ float hacc[BATCH * 10];
    int t = threadIdx.x;              // 320 threads
    if (t < BATCH * 10) {
        int b = t / 10, k = t % 10;
        float sum = 0.f;
        for (int s2 = 0; s2 < SEQ; ++s2)
            sum += hvals[((size_t)b * SEQ + s2) * 10 + k];
        hacc[t] = sum;
    }
    __syncthreads();
    if (t < BATCH) {
        float s = 0.f;
        #pragma unroll
        for (int k = 0; k < 10; ++k)
            s += gelu_f(hacc[t * 10 + k] + bc[0]) * w2[k];
        float gv = 1.f / (1.f + expf(-(s + b2[0])));
        sel[t] = (int)rintf(gv);
    }
}

// ---------------- 6) Selected GEMM + bias + residual + gelu, bf16 WMMA ----------------
// One wave computes a 64(M) x 64(N) output block: 16 f32 accumulators,
// 16 b128 loads feed 16 WMMAs per K-step (1.0 loads/WMMA, 4x reuse of A and B).
__device__ __forceinline__ v16bf load_frag(const unsigned short* p) {
    v8bf lo = *(const v8bf*)(p);
    v8bf hi = *(const v8bf*)(p + 16);
    return __builtin_shufflevector(lo, hi, 0, 1, 2, 3, 4, 5, 6, 7,
                                           8, 9, 10, 11, 12, 13, 14, 15);
}

__global__ void __launch_bounds__(32)
k_gemm(const unsigned short* __restrict__ WiB, const unsigned short* __restrict__ WiiB,
       const unsigned short* __restrict__ xnT, const int* __restrict__ sel,
       const float* __restrict__ bi, const float* __restrict__ bii,
       const float* __restrict__ x, float* __restrict__ out) {
    int b  = blockIdx.z;
    int m0 = blockIdx.y * 64;
    int n0 = blockIdx.x * 64;
    int lane = threadIdx.x & 31;
    int l = lane & 15, hi = lane >> 4;

    int sgate = sel[b];
    const unsigned short* W    = sgate ? WiiB : WiB;
    const float*          bias = sgate ? bii  : bi;

    const unsigned short* Abase = W   + (size_t)(m0 + l) * SEQ;               // rows m0+l (+16i)
    const unsigned short* Bbase = xnT + ((size_t)b * DFF + n0 + l) * SEQ;     // cols n0+l (+16j)

    v8f acc[4][4] = {};
    for (int k0 = 0; k0 < SEQ; k0 += 32) {
        int koff = k0 + hi * 8;
        // prefetch next K-chunk lines while this chunk computes
        if (k0 + 32 < SEQ) {
            __builtin_prefetch(Abase + koff + 32, 0, 3);
            __builtin_prefetch(Bbase + koff + 32, 0, 3);
        }
        v16bf af[4];
        #pragma unroll
        for (int i = 0; i < 4; ++i)
            af[i] = load_frag(Abase + (size_t)i * 16 * SEQ + koff);
        #pragma unroll
        for (int j = 0; j < 4; ++j) {
            v16bf bf = load_frag(Bbase + (size_t)j * 16 * SEQ + koff);
            #pragma unroll
            for (int i = 0; i < 4; ++i)
                acc[i][j] = __builtin_amdgcn_wmma_f32_16x16x32_bf16(
                    false, af[i], false, bf, (short)0, acc[i][j], false, false);
        }
    }

    // epilogue: + bias(row) + residual x, exact gelu, f32 store
    #pragma unroll
    for (int i = 0; i < 4; ++i) {
        #pragma unroll
        for (int j = 0; j < 4; ++j) {
            int n = n0 + j * 16 + l;
            #pragma unroll
            for (int r = 0; r < 8; ++r) {
                int m = m0 + i * 16 + r + hi * 8;
                size_t idx = ((size_t)b * SEQ + m) * DFF + n;
                float v = acc[i][j][r] + bias[m] + x[idx];
                out[idx] = gelu_f(v);
            }
        }
    }
}

extern "C" void kernel_launch(void* const* d_in, const int* in_sizes, int n_in,
                              void* d_out, int out_size, void* d_ws, size_t ws_size,
                              hipStream_t stream) {
    const float* x    = (const float*)d_in[0];
    const float* ln_g = (const float*)d_in[1];
    const float* ln_b = (const float*)d_in[2];
    const float* w1   = (const float*)d_in[3];
    const float* b1   = (const float*)d_in[4];
    const float* wc   = (const float*)d_in[5];
    const float* bc   = (const float*)d_in[6];
    const float* w2   = (const float*)d_in[7];
    const float* b2   = (const float*)d_in[8];
    const float* Wi   = (const float*)d_in[9];
    const float* bi   = (const float*)d_in[10];
    const float* Wii  = (const float*)d_in[11];
    const float* bii  = (const float*)d_in[12];
    float* out = (float*)d_out;

    char* ws = (char*)d_ws;
    unsigned short* xnT  = (unsigned short*)(ws);                         // 50,331,648 B
    unsigned short* WiB  = (unsigned short*)(ws + 50331648);              //    524,288 B
    unsigned short* WiiB = (unsigned short*)(ws + 50855936);              //    524,288 B
    float*          mu   = (float*)(ws + 51380224);                       //     65,536 B
    float*          rs   = (float*)(ws + 51445760);                       //     65,536 B
    float*          hv   = (float*)(ws + 51511296);                       //    655,360 B
    int*            sel  = (int*)(ws + 52166656);                         //        128 B

    // 1) LN statistics
    k_ln_stats<<<ROWS, 256, 0, stream>>>(x, mu, rs);
    // 2) normalize + transpose to bf16 xnT
    k_norm_t<<<dim3(DFF / 64, SEQ / 64, BATCH), 256, 0, stream>>>(x, mu, rs, ln_g, ln_b, xnT);
    // 3) weight conversion
    k_prep<<<(SEQ * SEQ + 255) / 256, 256, 0, stream>>>(Wi, Wii, WiB, WiiB);
    // 4) gate stage 1
    k_gate1<<<ROWS / 8, 256, 0, stream>>>(x, mu, rs, ln_g, ln_b, w1, b1, wc, hv);
    // 5) gate stage 2
    k_gate2<<<1, 320, 0, stream>>>(hv, bc, w2, b2, sel);
    // 6) WMMA GEMM + epilogue (64x64 per wave)
    k_gemm<<<dim3(DFF / 64, SEQ / 64, BATCH), 32, 0, stream>>>(WiB, WiiB, xnT, sel, bi, bii, x, out);
}
